// LinkPredictor_15994458211451
// MI455X (gfx1250) — compile-verified, matrix operand order checked by reading, not run on hardware
//
#include <hip/hip_runtime.h>

typedef __attribute__((ext_vector_type(16))) _Float16 v16h;
typedef __attribute__((ext_vector_type(8)))  _Float16 v8h;
typedef __attribute__((ext_vector_type(4)))  _Float16 v4h;
typedef __attribute__((ext_vector_type(8)))  float    v8f;

#define HDIM  128
#define H2DIM 64
#define MT    64    // edges per workgroup (4 M-tiles of 16)
#define SA  520     // edge_repr row stride in halves (512 + 8 pad -> distinct LDS banks)
#define SH1 136     // h1 row stride in halves (128 + 8)
#define SH2 68      // h2 row stride in floats (64 + 4)

union V16U { v16h v; v8h h[2]; };

// |x| via inline asm: opaque to InstCombine/SLP, so the compiler can never
// canonicalize it to the (bitcast->and 0x7fffffff->bitcast) fabs idiom that
// crashes gfx1250 ISel when vectorized.
__device__ __forceinline__ float abs_asm(float x) {
  float r;
  asm("v_max_num_f32 %0, %1, -%1" : "=v"(r) : "v"(x));
  return r;
}

__global__ __launch_bounds__(256) void link_pred_wmma_kernel(
    const float* __restrict__ z,
    const long long* __restrict__ eidx,
    const float* __restrict__ W1, const float* __restrict__ B1,
    const float* __restrict__ W2, const float* __restrict__ B2,
    const float* __restrict__ W3, const float* __restrict__ B3,
    float* __restrict__ out, int E)
{
  __shared__ _Float16 s_erepr[MT * SA];   // 64 x 512 f16 edge representation (66.5 KB)
  __shared__ _Float16 s_h1[MT * SH1];     // 64 x 128 f16 hidden 1 (17 KB)
  __shared__ float    s_h2[MT * SH2];     // 64 x 64  f32 hidden 2 (17 KB)
  __shared__ int      s_idx[2 * MT];      // 64 src + 64 dst node ids

  const int tid   = threadIdx.x;
  const int lane  = tid & 31;
  const int wave  = tid >> 5;
  const int mlane = lane & 15;
  const int hi8   = (lane >> 4) << 3;     // 0 for lanes 0-15, 8 for lanes 16-31

  // ---- phase 0: edge endpoint indices (clamped, uniform control flow) ----
  if (tid < 2 * MT) {
    long long e = (long long)blockIdx.x * MT + (tid & (MT - 1));
    if (e >= E) e = E - 1;
    long long idx = (tid < MT) ? eidx[e] : eidx[(long long)E + e];
    s_idx[tid] = (int)idx;
  }
  __syncthreads();

  // ---- phase 1: gather z rows, build [src|dst|src*dst||src-dst|] f16 panel ----
  #pragma unroll
  for (int r = 0; r < (MT * 32) / 256; ++r) {
    int c   = r * 256 + tid;        // (edge row 0..63) x (quad 0..31)
    int row = c >> 5;
    int col = (c & 31) * 4;
    const float4 s4 = *(const float4*)(z + (long long)s_idx[row]      * HDIM + col);
    const float4 d4 = *(const float4*)(z + (long long)s_idx[MT + row] * HDIM + col);
    _Float16* er = &s_erepr[row * SA + col];
    v4h hs = { (_Float16)s4.x, (_Float16)s4.y, (_Float16)s4.z, (_Float16)s4.w };
    v4h hd = { (_Float16)d4.x, (_Float16)d4.y, (_Float16)d4.z, (_Float16)d4.w };
    v4h hp = { (_Float16)(s4.x*d4.x), (_Float16)(s4.y*d4.y),
               (_Float16)(s4.z*d4.z), (_Float16)(s4.w*d4.w) };
    v4h ha = { (_Float16)abs_asm(s4.x-d4.x), (_Float16)abs_asm(s4.y-d4.y),
               (_Float16)abs_asm(s4.z-d4.z), (_Float16)abs_asm(s4.w-d4.w) };
    *(v4h*)(er)       = hs;
    *(v4h*)(er + 128) = hd;
    *(v4h*)(er + 256) = hp;
    *(v4h*)(er + 384) = ha;
  }
  __syncthreads();

  // ---- layer 1: [64x512]f16 @ W1[512x128] -> relu -> s_h1 ----
  // Each wave owns one 16-col N-tile; each B fragment feeds 4 M-tile WMMAs,
  // amortizing W1 (L2) traffic 4x versus a 16-edge tile.
  {
    const int nl = wave * 16 + mlane;            // output column
    v8f acc0 = {}, acc1 = {}, acc2 = {}, acc3 = {};
    for (int kk = 0; kk < 16; ++kk) {
      const int kb = kk * 32;
      // B fragment: W1 rows K, column nl; convert f32 (L2-resident) -> f16
      v16h b;
      #pragma unroll
      for (int j = 0; j < 16; ++j) {
        int K = kb + ((j & 8) << 1) + hi8 + (j & 7);
        b[j] = (_Float16)W1[K * HDIM + nl];
      }
      // A fragments: 16-bit 16x32 layout (lanes 0-15: K 0..7/16..23, 16-31: +8)
      #pragma unroll
      for (int mt = 0; mt < 4; ++mt) {
        V16U a;
        const _Float16* ar = &s_erepr[(mt * 16 + mlane) * SA + kb + hi8];
        a.h[0] = *(const v8h*)(ar);
        a.h[1] = *(const v8h*)(ar + 16);
        v8f& acc = (mt == 0) ? acc0 : (mt == 1) ? acc1 : (mt == 2) ? acc2 : acc3;
        acc = __builtin_amdgcn_wmma_f32_16x16x32_f16(false, a.v, false, b,
                                                     (short)0, acc, false, false);
      }
    }
    const float bv = B1[nl];
    #pragma unroll
    for (int mt = 0; mt < 4; ++mt) {
      v8f& acc = (mt == 0) ? acc0 : (mt == 1) ? acc1 : (mt == 2) ? acc2 : acc3;
      #pragma unroll
      for (int g = 0; g < 8; ++g) {              // C layout: row = g + hi8
        float v = acc[g] + bv;
        v = v > 0.f ? v : 0.f;
        s_h1[(mt * 16 + g + hi8) * SH1 + nl] = (_Float16)v;
      }
    }
  }
  __syncthreads();

  // ---- layer 2: [64x128]f16 @ W2[128x64] -> relu -> s_h2 ----
  // 16 (m,n) tiles over 8 waves: wave -> n-tile (wave&3), m-tiles 2*(wave>>2)+{0,1}
  {
    const int nt  = wave & 3;
    const int mt0 = (wave >> 2) * 2;
    const int nl  = nt * 16 + mlane;
    v8f acc0 = {}, acc1 = {};
    #pragma unroll
    for (int kk = 0; kk < 4; ++kk) {
      const int kb = kk * 32;
      v16h b;
      #pragma unroll
      for (int j = 0; j < 16; ++j) {
        int K = kb + ((j & 8) << 1) + hi8 + (j & 7);
        b[j] = (_Float16)W2[K * H2DIM + nl];
      }
      #pragma unroll
      for (int i = 0; i < 2; ++i) {
        V16U a;
        const _Float16* ar = &s_h1[((mt0 + i) * 16 + mlane) * SH1 + kb + hi8];
        a.h[0] = *(const v8h*)(ar);
        a.h[1] = *(const v8h*)(ar + 16);
        v8f& acc = (i == 0) ? acc0 : acc1;
        acc = __builtin_amdgcn_wmma_f32_16x16x32_f16(false, a.v, false, b,
                                                     (short)0, acc, false, false);
      }
    }
    const float bv = B2[nl];
    #pragma unroll
    for (int i = 0; i < 2; ++i) {
      v8f& acc = (i == 0) ? acc0 : acc1;
      #pragma unroll
      for (int g = 0; g < 8; ++g) {
        float v = acc[g] + bv;
        v = v > 0.f ? v : 0.f;
        s_h2[((mt0 + i) * 16 + g + hi8) * SH2 + nl] = v;
      }
    }
  }
  __syncthreads();

  // ---- layer 3: score[m] = h2[m,:] . W3 + b3 (64 lanes, trivial dot) ----
  if (tid < MT) {
    float sum = B3[0];
    #pragma unroll 8
    for (int c = 0; c < 64; ++c) sum += s_h2[tid * SH2 + c] * W3[c];
    long long e = (long long)blockIdx.x * MT + tid;
    if (e < E) out[e] = sum;
  }
}

extern "C" void kernel_launch(void* const* d_in, const int* in_sizes, int n_in,
                              void* d_out, int out_size, void* d_ws, size_t ws_size,
                              hipStream_t stream) {
  const float*     z    = (const float*)d_in[0];
  const long long* eidx = (const long long*)d_in[1];   // int64 edge_index [2, E]
  const float* W1 = (const float*)d_in[2];
  const float* B1 = (const float*)d_in[3];
  const float* W2 = (const float*)d_in[4];
  const float* B2 = (const float*)d_in[5];
  const float* W3 = (const float*)d_in[6];
  const float* B3 = (const float*)d_in[7];
  float* out = (float*)d_out;

  const int E = in_sizes[1] / 2;
  const int blocks = (E + MT - 1) / MT;
  link_pred_wmma_kernel<<<blocks, 256, 0, stream>>>(z, eidx, W1, B1, W2, B2,
                                                    W3, B3, out, E);
}
